// SetTransformerConditionEncoder_70987219468673
// MI455X (gfx1250) — matrix operand order repaired and checked by reading, not compiled, not run on hardware
//
#include <hip/hip_runtime.h>

// ---------------------------------------------------------------------------
// Set Transformer encoder for MI455X (gfx1250, wave32, WMMA).
// fp16 activations + weights, fp32 accumulation via v_wmma_f32_16x16x32_f16.
// ---------------------------------------------------------------------------

typedef _Float16 h16;
typedef __attribute__((ext_vector_type(16))) _Float16 v16h;
typedef __attribute__((ext_vector_type(8)))  float    v8f;

#define BB   64
#define NN   2048
#define DHD  512
#define NHH  8
#define NII  64
#define NSS  32

// ---------------- WMMA helpers (CDNA5 fragment layouts, ISA 7.12.2) --------

__device__ __forceinline__ v8f wmma32(v16h a, v16h b, v8f c) {
  return __builtin_amdgcn_wmma_f32_16x16x32_f16(
      /*neg_a=*/false, a, /*neg_b=*/false, b,
      /*c_mod=*/(short)0, c, /*reuse_a=*/false, /*reuse_b=*/false);
}

// A fragment: 16x32 f16.  lane = m + kmid*16 ; half idx h -> k = kmid*16 + h
__device__ __forceinline__ v16h load_a_frag(const h16* As, int stride, int m0, int k0, int lane) {
  v16h a;
  int m = lane & 15, kmid = (lane >> 4) << 4;
#pragma unroll
  for (int h = 0; h < 16; ++h) {
    int k = kmid + h;
    a[h] = As[(m0 + m) * stride + k0 + k];
  }
  return a;
}

// B fragment: 32x16 f16 (row-major K x N in LDS). lanes 0-15: k=0..15, lanes 16-31: k=16..31
__device__ __forceinline__ v16h load_b_frag(const h16* Bs, int stride, int k0, int n0, int lane) {
  v16h b;
  int n = lane & 15, kb = (lane >> 4) << 4;
#pragma unroll
  for (int h = 0; h < 16; ++h)
    b[h] = Bs[(k0 + kb + h) * stride + n0 + n];
  return b;
}

// B fragment from a transposed source: B(k,n) = Src[n][k]  (used for Q*K^T)
__device__ __forceinline__ v16h load_bt_frag(const h16* Src, int stride, int k0, int n0, int lane) {
  v16h b;
  int n = lane & 15, kb = (lane >> 4) << 4;
#pragma unroll
  for (int h = 0; h < 16; ++h)
    b[h] = Src[(n0 + n) * stride + k0 + kb + h];
  return b;
}

__device__ __forceinline__ float redmax16(float v) {
#pragma unroll
  for (int o = 8; o; o >>= 1) v = fmaxf(v, __shfl_xor(v, o, 16));
  return v;
}
__device__ __forceinline__ float redsum16(float v) {
#pragma unroll
  for (int o = 8; o; o >>= 1) v += __shfl_xor(v, o, 16);
  return v;
}

// ---------------- weight conversion ----------------------------------------

__global__ void f32_to_h_kernel(const float* __restrict__ s, h16* __restrict__ d, int n) {
  int i = blockIdx.x * 256 + threadIdx.x;
  if (i < n) d[i] = (h16)s[i];
}

// ---------------- input projection + exact gelu ----------------------------

__global__ __launch_bounds__(256)
void proj_in_kernel(const float* __restrict__ xv, const float* __restrict__ res,
                    const float* __restrict__ wp, const float* __restrict__ bp,
                    h16* __restrict__ X) {
  int row = blockIdx.x;
  int t = threadIdx.x;
  __shared__ float f[8];
  if (t < 6) f[t] = xv[(size_t)row * 6 + t];
  if (t == 6) f[6] = res[row];
  __syncthreads();
#pragma unroll
  for (int j = 0; j < 2; ++j) {
    int col = t + j * 256;
    float acc = bp[col];
#pragma unroll
    for (int k = 0; k < 7; ++k) acc += f[k] * wp[k * DHD + col];
    float g = 0.5f * acc * (1.0f + erff(acc * 0.70710678118f));
    X[(size_t)row * DHD + col] = (h16)g;
  }
}

// ---------------- generic 512x512 GEMM, fp16 in, fp16 out, WMMA ------------
// out[M x 512] = act(A[M x 512] @ W[512 x 512] + bias).  ACT: 0 none, 1 relu.

template <int ACT>
__global__ __launch_bounds__(256)
void gemm512(const h16* __restrict__ A, const h16* __restrict__ W,
             const float* __restrict__ bias, h16* __restrict__ out, int M) {
  __shared__ h16 As[64 * 40];
  __shared__ h16 Bs[32 * 72];
  int tid = threadIdx.x, lane = tid & 31, w = tid >> 5;
  int wm = w & 3, wn = w >> 2;                 // 4 (M) x 2 (N) waves
  int m0b = blockIdx.x * 64, n0b = blockIdx.y * 64;
  v8f acc0 = {}, acc1 = {};

  for (int k0 = 0; k0 < 512; k0 += 32) {
    { // stage A tile 64x32
      int r = tid >> 2, c = (tid & 3) * 8;
      int ar = m0b + r; if (ar >= M) ar = M - 1;
      uint4 v = *(const uint4*)(A + (size_t)ar * 512 + k0 + c);
      *(uint4*)&As[r * 40 + c] = v;
    }
    { // stage W tile 32x64
      int r = tid >> 3, c = (tid & 7) * 8;
      uint4 v = *(const uint4*)(W + (size_t)(k0 + r) * 512 + n0b + c);
      *(uint4*)&Bs[r * 72 + c] = v;
    }
    if (k0 + 32 < 512) // prefetch next A tile cacheline
      __builtin_prefetch(A + (size_t)(m0b + (tid >> 2)) * 512 + k0 + 32, 0, 1);
    __syncthreads();

    v16h a  = load_a_frag(As, 40, wm * 16, 0, lane);
    v16h b0 = load_b_frag(Bs, 72, 0, wn * 32 + 0, lane);
    v16h b1 = load_b_frag(Bs, 72, 0, wn * 32 + 16, lane);
    acc0 = wmma32(a, b0, acc0);
    acc1 = wmma32(a, b1, acc1);
    __syncthreads();
  }

  int col0 = n0b + wn * 32 + (lane & 15);
  int rbase = m0b + wm * 16 + ((lane >> 4) << 3);
#pragma unroll
  for (int r = 0; r < 8; ++r) {
    int row = rbase + r;
    if (row < M) {
      float v0 = acc0[r] + bias[col0];
      float v1 = acc1[r] + bias[col0 + 16];
      if (ACT == 1) { v0 = fmaxf(v0, 0.f); v1 = fmaxf(v1, 0.f); }
      out[(size_t)row * 512 + col0]      = (h16)v0;
      out[(size_t)row * 512 + col0 + 16] = (h16)v1;
    }
  }
}

// ---------------- attention, big-K (MAB0 / PMA): shared Q, nk = 2048 -------
// grid.x = B*NH, block = 32*(nq/16).  Streaming online softmax, 32 keys/iter.
// O[b*nq + q, h*64 + d] = Qh + softmax(mask(Q K^T / sqrt(512))) @ Vh

__global__ __launch_bounds__(128)
void attn_bigK(const h16* __restrict__ Qp, const h16* __restrict__ Kp,
               const h16* __restrict__ Vp, const int* __restrict__ mask,
               h16* __restrict__ O, int nq) {
  __shared__ h16 Qs[64 * 72];
  __shared__ h16 Ks[32 * 72];
  __shared__ h16 Vs[32 * 72];
  __shared__ h16 Ps[4][16 * 40];
  __shared__ float mf[32];

  int tid = threadIdx.x, lane = tid & 31, w = tid >> 5;
  int bh = blockIdx.x, b = bh >> 3, h = bh & 7;
  int hoff = h * 64;

  for (int i = tid; i < nq * 8; i += blockDim.x) {
    int r = i >> 3, c = (i & 7) * 8;
    *(uint4*)&Qs[r * 72 + c] = *(const uint4*)(Qp + (size_t)r * 512 + hoff + c);
  }

  v8f o0 = {}, o1 = {}, o2 = {}, o3 = {};
  float mst[8], lst[8];
#pragma unroll
  for (int r = 0; r < 8; ++r) { mst[r] = -1e30f; lst[r] = 0.f; }
  const float scale = 0.04419417382f; // 1/sqrt(512)

  for (int kb = 0; kb < NN; kb += 32) {
    __syncthreads();
    for (int i = tid; i < 32 * 8; i += blockDim.x) {
      int r = i >> 3, c = (i & 7) * 8;
      size_t g = ((size_t)b * NN + kb + r) * 512 + hoff + c;
      *(uint4*)&Ks[r * 72 + c] = *(const uint4*)(Kp + g);
      *(uint4*)&Vs[r * 72 + c] = *(const uint4*)(Vp + g);
    }
    if (tid < 32) mf[tid] = (float)mask[(size_t)b * NN + kb + tid];
    __syncthreads();

    // S = Q K^T : 16 rows x 32 keys per wave, contraction over d = 64
    v16h aQ0 = load_a_frag(Qs, 72, w * 16, 0, lane);
    v16h aQ1 = load_a_frag(Qs, 72, w * 16, 32, lane);
    v8f s0 = {}, s1 = {};
    s0 = wmma32(aQ0, load_bt_frag(Ks, 72, 0,  0, lane), s0);
    s0 = wmma32(aQ1, load_bt_frag(Ks, 72, 32, 0, lane), s0);
    s1 = wmma32(aQ0, load_bt_frag(Ks, 72, 0,  16, lane), s1);
    s1 = wmma32(aQ1, load_bt_frag(Ks, 72, 32, 16, lane), s1);

    float mk0 = mf[lane & 15], mk1 = mf[16 + (lane & 15)];
#pragma unroll
    for (int r = 0; r < 8; ++r) {
      float v0 = s0[r] * scale; if (mk0 == 0.f) v0 = -10000.f;
      float v1 = s1[r] * scale; if (mk1 == 0.f) v1 = -10000.f;
      float tm = redmax16(fmaxf(v0, v1));
      float mn = fmaxf(mst[r], tm);
      float fac = __expf(mst[r] - mn);
      float p0 = __expf(v0 - mn), p1 = __expf(v1 - mn);
      float rs = redsum16(p0 + p1);
      lst[r] = lst[r] * fac + rs;
      mst[r] = mn;
      o0[r] *= fac; o1[r] *= fac; o2[r] *= fac; o3[r] *= fac;
      s0[r] = p0; s1[r] = p1;
    }

    // stage P (C-layout -> A-layout via LDS)
    int pc = lane & 15, pr = (lane >> 4) << 3;
#pragma unroll
    for (int r = 0; r < 8; ++r) {
      Ps[w][(pr + r) * 40 + pc]      = (h16)s0[r];
      Ps[w][(pr + r) * 40 + pc + 16] = (h16)s1[r];
    }
    __syncthreads();

    v16h aP = load_a_frag(Ps[w], 40, 0, 0, lane);
    o0 = wmma32(aP, load_b_frag(Vs, 72, 0, 0,  lane), o0);
    o1 = wmma32(aP, load_b_frag(Vs, 72, 0, 16, lane), o1);
    o2 = wmma32(aP, load_b_frag(Vs, 72, 0, 32, lane), o2);
    o3 = wmma32(aP, load_b_frag(Vs, 72, 0, 48, lane), o3);
  }

  int col = lane & 15;
  int mloc = w * 16 + ((lane >> 4) << 3);
#pragma unroll
  for (int r = 0; r < 8; ++r) {
    int q = mloc + r;
    float inv = 1.0f / lst[r];
    size_t orow = ((size_t)b * nq + q) * 512 + hoff;
    O[orow + col]      = (h16)((float)Qs[q * 72 + col]      + o0[r] * inv);
    O[orow + col + 16] = (h16)((float)Qs[q * 72 + col + 16] + o1[r] * inv);
    O[orow + col + 32] = (h16)((float)Qs[q * 72 + col + 32] + o2[r] * inv);
    O[orow + col + 48] = (h16)((float)Qs[q * 72 + col + 48] + o3[r] * inv);
  }
}

// ---------------- attention, small-K (MAB1): per-batch Q (N rows), nk = 64 -
// grid = (B*NH, N/64), block 128.

__global__ __launch_bounds__(128)
void attn_smallK(const h16* __restrict__ Qp, const h16* __restrict__ Kp,
                 const h16* __restrict__ Vp, h16* __restrict__ O) {
  __shared__ h16 Qs[64 * 72];
  __shared__ h16 Ks[64 * 72];
  __shared__ h16 Vs[64 * 72];
  __shared__ h16 Ps[4][16 * 72];

  int tid = threadIdx.x, lane = tid & 31, w = tid >> 5;
  int bh = blockIdx.x, b = bh >> 3, h = bh & 7;
  int hoff = h * 64;
  size_t qbase = (size_t)b * NN + blockIdx.y * 64;

  for (int i = tid; i < 64 * 8; i += 128) {
    int r = i >> 3, c = (i & 7) * 8;
    *(uint4*)&Qs[r * 72 + c] = *(const uint4*)(Qp + (qbase + r) * 512 + hoff + c);
    size_t kg = ((size_t)b * NII + r) * 512 + hoff + c;
    *(uint4*)&Ks[r * 72 + c] = *(const uint4*)(Kp + kg);
    *(uint4*)&Vs[r * 72 + c] = *(const uint4*)(Vp + kg);
  }
  __syncthreads();

  v16h aQ0 = load_a_frag(Qs, 72, w * 16, 0, lane);
  v16h aQ1 = load_a_frag(Qs, 72, w * 16, 32, lane);
  v8f s0 = {}, s1 = {}, s2 = {}, s3 = {};
  s0 = wmma32(aQ0, load_bt_frag(Ks, 72, 0,  0,  lane), s0);
  s0 = wmma32(aQ1, load_bt_frag(Ks, 72, 32, 0,  lane), s0);
  s1 = wmma32(aQ0, load_bt_frag(Ks, 72, 0,  16, lane), s1);
  s1 = wmma32(aQ1, load_bt_frag(Ks, 72, 32, 16, lane), s1);
  s2 = wmma32(aQ0, load_bt_frag(Ks, 72, 0,  32, lane), s2);
  s2 = wmma32(aQ1, load_bt_frag(Ks, 72, 32, 32, lane), s2);
  s3 = wmma32(aQ0, load_bt_frag(Ks, 72, 0,  48, lane), s3);
  s3 = wmma32(aQ1, load_bt_frag(Ks, 72, 32, 48, lane), s3);

  const float scale = 0.04419417382f;
#pragma unroll
  for (int r = 0; r < 8; ++r) {
    float v0 = s0[r] * scale, v1 = s1[r] * scale;
    float v2 = s2[r] * scale, v3 = s3[r] * scale;
    float mx = redmax16(fmaxf(fmaxf(v0, v1), fmaxf(v2, v3)));
    float p0 = __expf(v0 - mx), p1 = __expf(v1 - mx);
    float p2 = __expf(v2 - mx), p3 = __expf(v3 - mx);
    float inv = 1.0f / redsum16(p0 + p1 + p2 + p3);
    s0[r] = p0 * inv; s1[r] = p1 * inv; s2[r] = p2 * inv; s3[r] = p3 * inv;
  }

  int pc = lane & 15, pr = (lane >> 4) << 3;
#pragma unroll
  for (int r = 0; r < 8; ++r) {
    Ps[w][(pr + r) * 72 + pc]      = (h16)s0[r];
    Ps[w][(pr + r) * 72 + pc + 16] = (h16)s1[r];
    Ps[w][(pr + r) * 72 + pc + 32] = (h16)s2[r];
    Ps[w][(pr + r) * 72 + pc + 48] = (h16)s3[r];
  }
  __syncthreads();

  v16h aP0 = load_a_frag(Ps[w], 72, 0, 0, lane);
  v16h aP1 = load_a_frag(Ps[w], 72, 0, 32, lane);
  v8f o0 = {}, o1 = {}, o2 = {}, o3 = {};
  o0 = wmma32(aP0, load_b_frag(Vs, 72, 0,  0,  lane), o0);
  o0 = wmma32(aP1, load_b_frag(Vs, 72, 32, 0,  lane), o0);
  o1 = wmma32(aP0, load_b_frag(Vs, 72, 0,  16, lane), o1);
  o1 = wmma32(aP1, load_b_frag(Vs, 72, 32, 16, lane), o1);
  o2 = wmma32(aP0, load_b_frag(Vs, 72, 0,  32, lane), o2);
  o2 = wmma32(aP1, load_b_frag(Vs, 72, 32, 32, lane), o2);
  o3 = wmma32(aP0, load_b_frag(Vs, 72, 0,  48, lane), o3);
  o3 = wmma32(aP1, load_b_frag(Vs, 72, 32, 48, lane), o3);

  int col = lane & 15;
  int mloc = w * 16 + ((lane >> 4) << 3);
#pragma unroll
  for (int r = 0; r < 8; ++r) {
    int q = mloc + r;
    size_t orow = (qbase + q) * 512 + hoff;
    O[orow + col]      = (h16)((float)Qs[q * 72 + col]      + o0[r]);
    O[orow + col + 16] = (h16)((float)Qs[q * 72 + col + 16] + o1[r]);
    O[orow + col + 32] = (h16)((float)Qs[q * 72 + col + 32] + o2[r]);
    O[orow + col + 48] = (h16)((float)Qs[q * 72 + col + 48] + o3[r]);
  }
}

// ---------------- layernorm kernels (wave per row, 512 cols) ---------------

__global__ __launch_bounds__(256)
void ln_kernel(const h16* __restrict__ X, const float* __restrict__ g,
               const float* __restrict__ bta, h16* __restrict__ out) {
  int w = threadIdx.x >> 5, lane = threadIdx.x & 31;
  size_t row = (size_t)blockIdx.x * 8 + w;
  const h16* xr = X + row * 512;
  float vals[16], s = 0.f, s2 = 0.f;
#pragma unroll
  for (int i = 0; i < 16; ++i) {
    float v = (float)xr[lane + i * 32];
    vals[i] = v; s += v; s2 += v * v;
  }
#pragma unroll
  for (int o = 16; o; o >>= 1) { s += __shfl_xor(s, o, 32); s2 += __shfl_xor(s2, o, 32); }
  float mean = s * (1.f / 512.f);
  float var = s2 * (1.f / 512.f) - mean * mean;
  float inv = rsqrtf(var + 1e-5f);
#pragma unroll
  for (int i = 0; i < 16; ++i) {
    int c = lane + i * 32;
    out[row * 512 + c] = (h16)((vals[i] - mean) * inv * g[c] + bta[c]);
  }
}

template <typename OutT>
__global__ __launch_bounds__(256)
void ln_res_kernel(const h16* __restrict__ T, const h16* __restrict__ G,
                   const float* __restrict__ g, const float* __restrict__ bta,
                   OutT* __restrict__ out) {
  int w = threadIdx.x >> 5, lane = threadIdx.x & 31;
  size_t row = (size_t)blockIdx.x * 8 + w;
  float vals[16], s = 0.f, s2 = 0.f;
#pragma unroll
  for (int i = 0; i < 16; ++i) {
    int c = lane + i * 32;
    float v = (float)T[row * 512 + c] + (float)G[row * 512 + c];
    vals[i] = v; s += v; s2 += v * v;
  }
#pragma unroll
  for (int o = 16; o; o >>= 1) { s += __shfl_xor(s, o, 32); s2 += __shfl_xor(s2, o, 32); }
  float mean = s * (1.f / 512.f);
  float var = s2 * (1.f / 512.f) - mean * mean;
  float inv = rsqrtf(var + 1e-5f);
#pragma unroll
  for (int i = 0; i < 16; ++i) {
    int c = lane + i * 32;
    out[row * 512 + c] = (OutT)((vals[i] - mean) * inv * g[c] + bta[c]);
  }
}

// ---------------------------------------------------------------------------
// Host orchestration
// ---------------------------------------------------------------------------

struct MabPtrs {
  const float *wq, *bq, *wk, *bk, *wv, *bv, *wo, *bo, *g0, *b0, *g1, *b1;
};
struct MabW { const h16 *wq, *wk, *wv, *wo; };

extern "C" void kernel_launch(void* const* d_in, const int* in_sizes, int n_in,
                              void* d_out, int out_size, void* d_ws, size_t ws_size,
                              hipStream_t stream) {
  (void)in_sizes; (void)n_in; (void)out_size; (void)ws_size;

  // ---- inputs, in setup_inputs() insertion order, nested depth-first ----
  int idx = 0;
  auto nf = [&]() { return (const float*)d_in[idx++]; };
  const float* x_values  = nf();
  const float* residuals = nf();
  const int*   point_mask = (const int*)d_in[idx++];
  const float* wp = nf();
  const float* bp = nf();
  auto readMab = [&]() {
    MabPtrs m;
    m.wq = nf(); m.bq = nf(); m.wk = nf(); m.bk = nf();
    m.wv = nf(); m.bv = nf(); m.wo = nf(); m.bo = nf();
    m.g0 = nf(); m.b0 = nf(); m.g1 = nf(); m.b1 = nf();
    return m;
  };
  struct Isab { const float* I; MabPtrs m0, m1; } isabs[3];
  for (int l = 0; l < 3; ++l) {
    isabs[l].I  = nf();
    isabs[l].m0 = readMab();
    isabs[l].m1 = readMab();
  }
  const float* Sseed = nf();
  MabPtrs pmam = readMab();

  // ---- workspace bump allocator (~690 MB total) ----
  size_t off = 0;
  auto bump = [&](size_t bytes) -> void* {
    void* p = (char*)d_ws + off;
    off += (bytes + 255) & ~(size_t)255;
    return p;
  };
  const size_t BN = (size_t)BB * NN;          // 131072
  h16* Xh = (h16*)bump(BN * DHD * sizeof(h16));
  h16* Qp = (h16*)bump(BN * DHD * sizeof(h16));
  h16* Kp = (h16*)bump(BN * DHD * sizeof(h16));
  h16* Vp = (h16*)bump(BN * DHD * sizeof(h16));
  h16* Ob = (h16*)bump(BN * DHD * sizeof(h16));
  h16* Hh = (h16*)bump((size_t)BB * NII * DHD * sizeof(h16));
  h16* Th = Kp;   // post-LN0 activations (Kp dead after attention)
  h16* Gh = Vp;   // relu(FFN) branch     (Vp dead after attention)

  auto conv = [&](const float* src, int n) -> h16* {
    h16* dst = (h16*)bump((size_t)n * sizeof(h16));
    f32_to_h_kernel<<<dim3((n + 255) / 256), dim3(256), 0, stream>>>(src, dst, n);
    return dst;
  };
  MabW Wd[7];
  const h16* Ih[3]; const h16* Sh;
  {
    int wi = 0;
    for (int l = 0; l < 3; ++l) {
      Ih[l] = conv(isabs[l].I, NII * DHD);
      const MabPtrs* ms[2] = { &isabs[l].m0, &isabs[l].m1 };
      for (int j = 0; j < 2; ++j) {
        Wd[wi].wq = conv(ms[j]->wq, DHD * DHD);
        Wd[wi].wk = conv(ms[j]->wk, DHD * DHD);
        Wd[wi].wv = conv(ms[j]->wv, DHD * DHD);
        Wd[wi].wo = conv(ms[j]->wo, DHD * DHD);
        ++wi;
      }
    }
    Sh = conv(Sseed, NSS * DHD);
    Wd[6].wq = conv(pmam.wq, DHD * DHD);
    Wd[6].wk = conv(pmam.wk, DHD * DHD);
    Wd[6].wv = conv(pmam.wv, DHD * DHD);
    Wd[6].wo = conv(pmam.wo, DHD * DHD);
  }

  // ---- input projection + gelu ----
  proj_in_kernel<<<dim3((unsigned)BN), dim3(256), 0, stream>>>(
      x_values, residuals, wp, bp, Xh);

  auto gemm = [&](const h16* A, const h16* W, const float* bias, h16* out,
                  size_t M, int act) {
    dim3 g((unsigned)((M + 63) / 64), 8);
    if (act == 0) gemm512<0><<<g, dim3(256), 0, stream>>>(A, W, bias, out, (int)M);
    else          gemm512<1><<<g, dim3(256), 0, stream>>>(A, W, bias, out, (int)M);
  };
  auto epilogue = [&](const MabPtrs& P, const MabW& W, size_t Mo,
                      h16* outH, float* outF) {
    ln_kernel<<<dim3((unsigned)(Mo / 8)), dim3(256), 0, stream>>>(Ob, P.g0, P.b0, Th);
    gemm(Th, W.wo, P.bo, Gh, Mo, /*relu*/1);
    if (outF)
      ln_res_kernel<float><<<dim3((unsigned)(Mo / 8)), dim3(256), 0, stream>>>(
          Th, Gh, P.g1, P.b1, outF);
    else
      ln_res_kernel<h16><<<dim3((unsigned)(Mo / 8)), dim3(256), 0, stream>>>(
          Th, Gh, P.g1, P.b1, outH);
  };

  // ---- 3 ISAB layers ----
  for (int l = 0; l < 3; ++l) {
    const MabW& W0 = Wd[l * 2 + 0];
    const MabW& W1 = Wd[l * 2 + 1];
    // MAB0: H = MAB(I, X, mask)   (Q shared across batch)
    gemm(Ih[l], W0.wq, isabs[l].m0.bq, Qp, NII, 0);
    gemm(Xh,    W0.wk, isabs[l].m0.bk, Kp, BN,  0);
    gemm(Xh,    W0.wv, isabs[l].m0.bv, Vp, BN,  0);
    attn_bigK<<<dim3(BB * NHH), dim3(128), 0, stream>>>(
        Qp, Kp, Vp, point_mask, Ob, NII);
    epilogue(isabs[l].m0, W0, (size_t)BB * NII, Hh, nullptr);
    // MAB1: X = MAB(X, H, ones)
    gemm(Xh, W1.wq, isabs[l].m1.bq, Qp, BN, 0);
    gemm(Hh, W1.wk, isabs[l].m1.bk, Kp, (size_t)BB * NII, 0);
    gemm(Hh, W1.wv, isabs[l].m1.bv, Vp, (size_t)BB * NII, 0);
    attn_smallK<<<dim3(BB * NHH, NN / 64), dim3(128), 0, stream>>>(Qp, Kp, Vp, Ob);
    epilogue(isabs[l].m1, W1, BN, Xh, nullptr);
  }

  // ---- PMA: out = MAB(S, X, mask), fp32 to d_out ----
  gemm(Sh, Wd[6].wq, pmam.bq, Qp, NSS, 0);
  gemm(Xh, Wd[6].wk, pmam.bk, Kp, BN,  0);
  gemm(Xh, Wd[6].wv, pmam.bv, Vp, BN,  0);
  attn_bigK<<<dim3(BB * NHH), dim3(64), 0, stream>>>(
      Qp, Kp, Vp, point_mask, Ob, NSS);
  epilogue(pmam, Wd[6], (size_t)BB * NSS, nullptr, (float*)d_out);
}
